// AFTAttention_18657337934229
// MI455X (gfx1250) — compile-verified, matrix operand order checked by reading, not run on hardware
//
#include <hip/hip_runtime.h>

// ---------------------------------------------------------------------------
// AFT attention for MI455X (gfx1250, wave32, WMMA + async global->LDS staging)
//   Q = x@WqT + bq ; K = x@WkT + bk ; V = x@WvT + bv      (bf16 WMMA, f32 acc)
//   numer = exp(K + wbias); denom = sum_b numer; weighted = sum_b numer*V/denom
//   Y = sigmoid(Q) * weighted
// ---------------------------------------------------------------------------

typedef __bf16        v16bf __attribute__((ext_vector_type(16)));
typedef float         v8f   __attribute__((ext_vector_type(8)));
typedef unsigned int  v4u   __attribute__((ext_vector_type(4)));
typedef int           v4i   __attribute__((ext_vector_type(4)));

#define B_  8
#define T_  4096
#define D_  1024
#define H_  1024
#define M_  (B_ * T_)          // 32768 rows of the fused GEMM

// --- async global->LDS support (toolchain-dependent) -----------------------
#if defined(__has_builtin)
#  if __has_builtin(__builtin_amdgcn_global_load_async_to_lds_b128) && \
      __has_builtin(__builtin_amdgcn_s_wait_asynccnt)
#    define HAS_ASYNC 1
#  endif
#endif
#ifndef HAS_ASYNC
#  define HAS_ASYNC 0
#endif

// Builtin signature (from clang diagnostic): arg0 = v4i addrspace(1)* (global
// source), arg1 = LDS destination, then imm offset + imm cpol.
#define GPTR(p) ((__attribute__((address_space(1))) v4i*)(void*)(p))
#define LPTR(p) ((__attribute__((address_space(3))) v4i*)(p))

// LDS tile geometry: 128x32 A tile, 3x 64x32 W tiles, rows padded to 40
// elements (80 B) so 16-lane b128 fragment reads stripe all 64 LDS banks.
#define LP        40
#define A_ELEMS   (128 * LP)                    // 5120
#define W_ELEMS   (64 * LP)                     // 2560
#define A_OFF(b)  ((b) * A_ELEMS)
#define Q_OFF(b)  (2 * A_ELEMS + (b) * W_ELEMS)
#define K_OFF(b)  (2 * A_ELEMS + 2 * W_ELEMS + (b) * W_ELEMS)
#define V_OFF(b)  (2 * A_ELEMS + 4 * W_ELEMS + (b) * W_ELEMS)
#define LDS_TOT   (2 * A_ELEMS + 6 * W_ELEMS)   // 25600 u16 = 51200 B

union FragU { v16bf f; v4u u[2]; };

// fp32 -> bf16, round-to-nearest-even
__device__ __forceinline__ unsigned short f2bf(float x) {
    unsigned int u = __float_as_uint(x);
    u += 0x7FFFu + ((u >> 16) & 1u);
    return (unsigned short)(u >> 16);
}

__global__ __launch_bounds__(256) void cvt_f32_bf16(const float* __restrict__ in,
                                                    unsigned short* __restrict__ out,
                                                    int n4) {
    int i = blockIdx.x * blockDim.x + threadIdx.x;
    if (i < n4) {
        float4 f = ((const float4*)in)[i];
        ushort4 o;
        o.x = f2bf(f.x); o.y = f2bf(f.y); o.z = f2bf(f.z); o.w = f2bf(f.w);
        ((ushort4*)out)[i] = o;
    }
}

// Load one 16x32 bf16 fragment (A layout, ISA 7.12.2) from LDS (or global):
// lane L -> row (base + L%16), K-chunks at +8*(L/16) and +16+8*(L/16).
__device__ __forceinline__ v16bf load_frag(const unsigned short* __restrict__ p) {
    FragU fu;
    fu.u[0] = *(const v4u*)(p);
    fu.u[1] = *(const v4u*)(p + 16);
    return fu.f;
}

// Stage one k-tile (A: 128x32, Wq/Wk/Wv: 64x32) into LDS buffer `buf`.
// 512 A-chunks + 3*256 W-chunks of 16 B; each thread moves 2 A + 3 W chunks
// => exactly 5 async ops per thread per tile (ASYNCcnt += 5 per wave-slot).
__device__ __forceinline__ void copy_tile(
    const unsigned short* __restrict__ xg, const unsigned short* __restrict__ qg,
    const unsigned short* __restrict__ kg, const unsigned short* __restrict__ vg,
    unsigned short* lds, int buf, int mblk, int nblk, int k0, int tid)
{
#pragma unroll
    for (int j = 0; j < 2; ++j) {
        const int c   = tid + j * 256;
        const int row = c >> 2, q = c & 3;
        const unsigned short* g = xg + (size_t)(mblk + row) * D_ + k0 + q * 8;
        unsigned short*       l = lds + A_OFF(buf) + row * LP + q * 8;
#if HAS_ASYNC
        __builtin_amdgcn_global_load_async_to_lds_b128(GPTR(g), LPTR(l), 0, 0);
#else
        *(v4u*)l = *(const v4u*)g;
#endif
    }
    const int row = tid >> 2, q = tid & 3;
    const unsigned short* gq = qg + (size_t)(nblk + row) * D_ + k0 + q * 8;
    const unsigned short* gk = kg + (size_t)(nblk + row) * D_ + k0 + q * 8;
    const unsigned short* gv = vg + (size_t)(nblk + row) * D_ + k0 + q * 8;
    unsigned short* lq = lds + Q_OFF(buf) + row * LP + q * 8;
    unsigned short* lk = lds + K_OFF(buf) + row * LP + q * 8;
    unsigned short* lv = lds + V_OFF(buf) + row * LP + q * 8;
#if HAS_ASYNC
    __builtin_amdgcn_global_load_async_to_lds_b128(GPTR(gq), LPTR(lq), 0, 0);
    __builtin_amdgcn_global_load_async_to_lds_b128(GPTR(gk), LPTR(lk), 0, 0);
    __builtin_amdgcn_global_load_async_to_lds_b128(GPTR(gv), LPTR(lv), 0, 0);
#else
    *(v4u*)lq = *(const v4u*)gq;
    *(v4u*)lk = *(const v4u*)gk;
    *(v4u*)lv = *(const v4u*)gv;
#endif
}

__global__ __launch_bounds__(256) void qkv_gemm(
    const unsigned short* __restrict__ xb,   // [M_, D_] bf16
    const unsigned short* __restrict__ wqb,  // [H_, D_] bf16
    const unsigned short* __restrict__ wkb,
    const unsigned short* __restrict__ wvb,
    const float* __restrict__ bq, const float* __restrict__ bk,
    const float* __restrict__ bv, const float* __restrict__ wbias,
    float* __restrict__ qsig,                // d_out [M_, H_] (holds sigmoid(Q))
    float* __restrict__ denom,               // [T_, H_]
    float* __restrict__ wsum)                // [T_, H_]
{
    __shared__ unsigned short lds[LDS_TOT];

    const int tid  = threadIdx.x;
    const int lane = tid & 31;
    const int wave = tid >> 5;
    const int mw   = wave >> 2;              // 0..1
    const int nw   = wave & 3;               // 0..3
    const int r    = lane & 15;
    const int half = lane >> 4;

    const int mblk = blockIdx.x * 128;
    const int nblk = blockIdx.y * 64;
    const int NK   = D_ / 32;                // 32 k-tiles

    v8f accQ[4], accK[4], accV[4];
#pragma unroll
    for (int s = 0; s < 4; ++s) {
        accQ[s] = (v8f){0,0,0,0,0,0,0,0};
        accK[s] = (v8f){0,0,0,0,0,0,0,0};
        accV[s] = (v8f){0,0,0,0,0,0,0,0};
    }

    // Fragment base offsets inside an LDS tile (buffer offset added per iter).
    const int aIdx[4] = {
        (mw * 64 +  0 + r) * LP + half * 8,
        (mw * 64 + 16 + r) * LP + half * 8,
        (mw * 64 + 32 + r) * LP + half * 8,
        (mw * 64 + 48 + r) * LP + half * 8 };
    const int bIdx = (nw * 16 + r) * LP + half * 8;

#if HAS_ASYNC
    // Double-buffered pipeline: issue tiles 0 and 1 up front.
    copy_tile(xb, wqb, wkb, wvb, lds, 0, mblk, nblk, 0,  tid);
    copy_tile(xb, wqb, wkb, wvb, lds, 1, mblk, nblk, 32, tid);
#endif

    for (int kt = 0; kt < NK; ++kt) {
#if HAS_ASYNC
        const int b = kt & 1;
        if (kt + 1 < NK) __builtin_amdgcn_s_wait_asynccnt(5);  // tile kt resident
        else             __builtin_amdgcn_s_wait_asynccnt(0);
        __syncthreads();                                       // all waves' tile kt
#else
        const int b = 0;
        __syncthreads();
        copy_tile(xb, wqb, wkb, wvb, lds, 0, mblk, nblk, kt * 32, tid);
        __syncthreads();
#endif

        v16bf fq = load_frag(lds + Q_OFF(b) + bIdx);
        v16bf fk = load_frag(lds + K_OFF(b) + bIdx);
        v16bf fv = load_frag(lds + V_OFF(b) + bIdx);
        v16bf a[4];
#pragma unroll
        for (int s = 0; s < 4; ++s) a[s] = load_frag(lds + A_OFF(b) + aIdx[s]);

#pragma unroll
        for (int s = 0; s < 4; ++s) {
            accQ[s] = __builtin_amdgcn_wmma_f32_16x16x32_bf16(
                false, a[s], false, fq, (short)0, accQ[s], false, false);
            accK[s] = __builtin_amdgcn_wmma_f32_16x16x32_bf16(
                false, a[s], false, fk, (short)0, accK[s], false, false);
            accV[s] = __builtin_amdgcn_wmma_f32_16x16x32_bf16(
                false, a[s], false, fv, (short)0, accV[s], false, false);
        }

#if HAS_ASYNC
        __syncthreads();                                       // buf b free
        if (kt + 2 < NK)
            copy_tile(xb, wqb, wkb, wvb, lds, b, mblk, nblk, (kt + 2) * 32, tid);
#endif
    }

    // Epilogue. C/D layout: VGPR i, lanes 0-15 -> (M=i, N=lane); lanes 16-31 ->
    // (M=i+8, N=lane-16). So row_local = i + 8*half, col_local = r.
    const int   h   = nblk + nw * 16 + r;
    const float bqv = bq[h];
    const float bkw = bk[h] + wbias[h];
    const float bvv = bv[h];

#pragma unroll
    for (int s = 0; s < 4; ++s) {
        const int mbase = mblk + mw * 64 + s * 16 + half * 8;
#pragma unroll
        for (int i = 0; i < 8; ++i) {
            const int m = mbase + i;
            const int t = m & (T_ - 1);              // m = b*T + t
            const float q  = accQ[s][i] + bqv;
            qsig[(size_t)m * H_ + h] = 1.0f / (1.0f + __expf(-q));
            const float nu = __expf(accK[s][i] + bkw);
            const float vv = accV[s][i] + bvv;
            float* dptr = &denom[(size_t)t * H_ + h];
            float* wptr = &wsum [(size_t)t * H_ + h];
            __hip_atomic_fetch_add(dptr, nu,      __ATOMIC_RELAXED, __HIP_MEMORY_SCOPE_AGENT);
            __hip_atomic_fetch_add(wptr, nu * vv, __ATOMIC_RELAXED, __HIP_MEMORY_SCOPE_AGENT);
        }
    }
}

__global__ __launch_bounds__(256) void finalize(float* __restrict__ out,
                                                const float* __restrict__ denom,
                                                const float* __restrict__ wsum,
                                                int n) {
    int i = blockIdx.x * blockDim.x + threadIdx.x;
    if (i < n) {
        int th = i & (T_ * H_ - 1);   // idx mod (T*H) == t*H + h
        out[i] = out[i] * (wsum[th] / denom[th]);
    }
}

extern "C" void kernel_launch(void* const* d_in, const int* in_sizes, int n_in,
                              void* d_out, int out_size, void* d_ws, size_t ws_size,
                              hipStream_t stream) {
    const float* x     = (const float*)d_in[0];
    const float* Wq    = (const float*)d_in[1];
    const float* bq    = (const float*)d_in[2];
    const float* Wk    = (const float*)d_in[3];
    const float* bk    = (const float*)d_in[4];
    const float* Wv    = (const float*)d_in[5];
    const float* bv    = (const float*)d_in[6];
    const float* wbias = (const float*)d_in[7];
    float* out = (float*)d_out;

    // Workspace layout:
    //   [0,   64MB)  x in bf16            (M_*D_ u16)
    //   [64,  70MB)  Wq,Wk,Wv in bf16     (3*H_*D_ u16)
    //   [70,  86MB)  denom  [T_,H_] f32
    //   [86, 102MB)  wsum   [T_,H_] f32
    char* ws = (char*)d_ws;
    unsigned short* xb  = (unsigned short*)ws;
    unsigned short* wqb = (unsigned short*)(ws + (size_t)M_ * D_ * 2);
    unsigned short* wkb = wqb + (size_t)H_ * D_;
    unsigned short* wvb = wkb + (size_t)H_ * D_;
    float* denom = (float*)(ws + (size_t)M_ * D_ * 2 + (size_t)3 * H_ * D_ * 2);
    float* wsum  = denom + (size_t)T_ * H_;

    (void)hipMemsetAsync(denom, 0, (size_t)2 * T_ * H_ * sizeof(float), stream);

    const int nx4 = M_ * D_ / 4;
    cvt_f32_bf16<<<nx4 / 256, 256, 0, stream>>>(x, xb, nx4);
    const int nw4 = H_ * D_ / 4;
    cvt_f32_bf16<<<nw4 / 256, 256, 0, stream>>>(Wq, wqb, nw4);
    cvt_f32_bf16<<<nw4 / 256, 256, 0, stream>>>(Wk, wkb, nw4);
    cvt_f32_bf16<<<nw4 / 256, 256, 0, stream>>>(Wv, wvb, nw4);

    dim3 grid(M_ / 128, H_ / 64);   // 256 x 16 workgroups, 256 threads each
    qkv_gemm<<<grid, 256, 0, stream>>>(xb, wqb, wkb, wvb, bq, bk, bv, wbias,
                                       out, denom, wsum);

    const int n = M_ * H_;
    finalize<<<n / 256, 256, 0, stream>>>(out, denom, wsum, n);
}
